// MultiHeadedAttentionTopicAware_35235911696712
// MI455X (gfx1250) — matrix operand order repaired and checked by reading, not compiled, not run on hardware
//
#include <hip/hip_runtime.h>
#include <hip/hip_bf16.h>

// Topic-aware multi-head attention for MI455X (gfx1250, wave32, WMMA).
// Heavy GEMMs + attention score/PV matmuls use v_wmma_f32_16x16x32_bf16.
// GEMM A-operand is staged block-wide into LDS via GLOBAL_LOAD_ASYNC_TO_LDS_B128
// (ASYNCcnt) so all 8 waves share one copy of the A strip.

typedef __bf16 bf16;
typedef __attribute__((ext_vector_type(16))) __bf16 v16bf;
typedef __attribute__((ext_vector_type(8)))  __bf16 v8bf;
typedef __attribute__((ext_vector_type(8)))  float  v8f;

union FragU { v16bf v; v8bf h[2]; };

// ---- Fragment loaders (layouts per CDNA5 ISA §7.12.2) ----
// A-matrix 16x32 bf16: lane m=lane&15; lanes<16 hold K {0..7,16..23}, lanes>=16 hold K {8..15,24..31}.
__device__ __forceinline__ v16bf load_frag_a(const bf16* __restrict__ base, int ld, unsigned lane) {
    const int half = (lane >> 4) & 1;
    const int m    = lane & 15;
    const bf16* r  = base + (size_t)m * ld + half * 8;
    FragU f;
    f.h[0] = *(const v8bf*)(r);
    f.h[1] = *(const v8bf*)(r + 16);
    return f.v;
}
// B-matrix 32x16 bf16: lane column n=lane&15; lanes<16 hold K {0..7,16..23}, lanes>=16 hold K {8..15,24..31}.
// Memory holds B^T row-major (row n, K contiguous) -> two 16B reads per lane.
__device__ __forceinline__ v16bf load_frag_b(const bf16* __restrict__ base, int ld, unsigned lane) {
    const int half = (lane >> 4) & 1;
    const int n    = lane & 15;
    const bf16* r  = base + (size_t)n * ld + half * 8;
    FragU f;
    f.h[0] = *(const v8bf*)(r);
    f.h[1] = *(const v8bf*)(r + 16);
    return f.v;
}

// Async 16B copy global -> LDS (CDNA5 ISA, GVS mode).
// lds_addr: LDS byte address (low 32 bits of flat shared pointer);
// gbase: 64-bit SGPR base; goff: per-lane 32-bit byte offset.
__device__ __forceinline__ void async_copy_b128(unsigned lds_addr,
                                                unsigned long long gbase,
                                                unsigned goff) {
    asm volatile("global_load_async_to_lds_b128 %0, %1, %2"
                 :: "v"(lds_addr), "v"(goff), "s"(gbase)
                 : "memory");
}
__device__ __forceinline__ void wait_asynccnt0() {
    asm volatile("s_wait_asynccnt 0" ::: "memory");
}

// ---- Generic NT GEMM: out[m,n] = sum_k A[m,k]*W[n,k] + bias[n] ----
// Block: 256 threads (8 waves) computes 32(M) x 512(N).
// A strip (32 x K bf16) staged once in LDS via async copies; each wave then
// computes a 32x64 tile: 2 A-frags (ds_load) x 4 B-frags (global) -> 8 WMMA/k-step.
// grid = (M/32, N/512), dyn-LDS = 32*K*2 bytes.
__global__ __launch_bounds__(256) void gemm_nt_bf16(
    const bf16* __restrict__ A, const bf16* __restrict__ W,
    const float* __restrict__ bias,
    float* __restrict__ outF, bf16* __restrict__ outH, float hscale,
    int M, int N, int K)
{
    extern __shared__ char smem_raw[];
    bf16* As = (bf16*)smem_raw;                       // [32][K]

    const unsigned tid  = threadIdx.x;
    const unsigned lane = tid & 31u;
    const unsigned wave = tid >> 5;
    const int m0 = blockIdx.x * 32;
    const int n0 = (blockIdx.y * 8 + wave) * 64;

    // ---- stage A strip into LDS (coalesced 16B async copies) ----
    {
        const int nchunk = 4 * K;                     // 32 rows * K * 2B / 16B
        const unsigned lds_base = (unsigned)(size_t)As;
        const unsigned long long gbase = (unsigned long long)(size_t)(A + (size_t)m0 * K);
        for (int c = tid; c < nchunk; c += 256)
            async_copy_b128(lds_base + (unsigned)c * 16u, gbase, (unsigned)c * 16u);
        wait_asynccnt0();
    }
    __syncthreads();

    v8f acc[8] = {};                                  // [2 m-tiles][4 n-tiles]
    for (int kb = 0; kb < K; kb += 32) {
        v16bf a0 = load_frag_a(As + kb,          K, lane);
        v16bf a1 = load_frag_a(As + 16 * K + kb, K, lane);
        v16bf b0 = load_frag_b(W + (size_t)(n0 +  0) * K + kb, K, lane);
        v16bf b1 = load_frag_b(W + (size_t)(n0 + 16) * K + kb, K, lane);
        v16bf b2 = load_frag_b(W + (size_t)(n0 + 32) * K + kb, K, lane);
        v16bf b3 = load_frag_b(W + (size_t)(n0 + 48) * K + kb, K, lane);
        acc[0] = __builtin_amdgcn_wmma_f32_16x16x32_bf16(false, a0, false, b0, (short)0, acc[0], false, false);
        acc[1] = __builtin_amdgcn_wmma_f32_16x16x32_bf16(false, a0, false, b1, (short)0, acc[1], false, false);
        acc[2] = __builtin_amdgcn_wmma_f32_16x16x32_bf16(false, a0, false, b2, (short)0, acc[2], false, false);
        acc[3] = __builtin_amdgcn_wmma_f32_16x16x32_bf16(false, a0, false, b3, (short)0, acc[3], false, false);
        acc[4] = __builtin_amdgcn_wmma_f32_16x16x32_bf16(false, a1, false, b0, (short)0, acc[4], false, false);
        acc[5] = __builtin_amdgcn_wmma_f32_16x16x32_bf16(false, a1, false, b1, (short)0, acc[5], false, false);
        acc[6] = __builtin_amdgcn_wmma_f32_16x16x32_bf16(false, a1, false, b2, (short)0, acc[6], false, false);
        acc[7] = __builtin_amdgcn_wmma_f32_16x16x32_bf16(false, a1, false, b3, (short)0, acc[7], false, false);
    }

    const int half = (lane >> 4) & 1;
    const int nn   = lane & 15;
#pragma unroll
    for (int mt = 0; mt < 2; ++mt) {
#pragma unroll
        for (int j = 0; j < 4; ++j) {
            const int n  = n0 + j * 16 + nn;
            const float bz = bias ? bias[n] : 0.0f;
            const v8f c = acc[mt * 4 + j];
#pragma unroll
            for (int r = 0; r < 8; ++r) {
                const int m = m0 + mt * 16 + r + half * 8;
                const float val = c[r] + bz;
                if (outF) outF[(size_t)m * N + n] = val;
                if (outH) outH[(size_t)m * N + n] = (bf16)(val * hscale);
            }
        }
    }
}

// ---- Conversion / pad / transpose helpers ----
__global__ void cvt_f32_bf16(const float* __restrict__ src, bf16* __restrict__ dst, int n) {
    int i = blockIdx.x * 256 + threadIdx.x;
    if (i < n) dst[i] = (bf16)src[i];
}
__global__ void pad_cvt(const float* __restrict__ src, bf16* __restrict__ dst,
                        int rows, int incols, int outcols) {
    int i = blockIdx.x * 256 + threadIdx.x;
    if (i >= rows * outcols) return;
    int r = i / outcols, c = i - r * outcols;
    dst[i] = (c < incols) ? (bf16)src[(size_t)r * incols + c] : (bf16)0.0f;
}
// v_s: [B,L,H*DH] bf16  ->  vT: [B,H,DH,L] bf16 (contiguous K=L for B-fragments of P@V)
__global__ void transpose_v(const bf16* __restrict__ v, bf16* __restrict__ vT) {
    int i = blockIdx.x * 256 + threadIdx.x;
    if (i >= 4 * 16 * 64 * 1024) return;
    int l = i & 1023;
    int d = (i >> 10) & 63;
    int h = (i >> 16) & 15;
    int b = i >> 20;
    vT[i] = v[(((size_t)b * 1024) + l) * 1024 + h * 64 + d];
}

// ---- Gate: sigmoid([q_lin|tk_lin|tv_lin] @ Wtw^T + btw), [4096 x 16] ----
__global__ __launch_bounds__(256) void gate_kernel(
    const float* __restrict__ ql, const float* __restrict__ tkl, const float* __restrict__ tvl,
    const float* __restrict__ Wtw, const float* __restrict__ btw, float* __restrict__ gate)
{
    int idx = blockIdx.x * 256 + threadIdx.x;
    if (idx >= 4096 * 16) return;
    int n = idx >> 4, h = idx & 15;
    const float* w = Wtw + (size_t)h * 3072;
    const float* a = ql  + (size_t)n * 1024;
    const float* b = tkl + (size_t)n * 1024;
    const float* c = tvl + (size_t)n * 1024;
    float s = btw[h];
    for (int d = 0; d < 1024; ++d) s += a[d] * w[d];
    for (int d = 0; d < 1024; ++d) s += b[d] * w[1024 + d];
    for (int d = 0; d < 1024; ++d) s += c[d] * w[2048 + d];
    gate[idx] = 1.0f / (1.0f + __expf(-s));
}

// ---- Fused attention per (b, h, 16-query tile) ----
// block = 128 threads (4 waves); dyn-LDS = 16x1024 f32 scores + stats.
// Scale 1/sqrt(DH) pre-folded into q_s/tq_s; softmax normalization folded into final scale.
__global__ __launch_bounds__(128) void attn_kernel(
    const bf16* __restrict__ qs,  const bf16* __restrict__ ks,
    const bf16* __restrict__ tqs, const bf16* __restrict__ tks,
    const bf16* __restrict__ vT,  const float* __restrict__ gate,
    const unsigned char* __restrict__ mask, bf16* __restrict__ ctx)
{
    extern __shared__ float smem[];
    float* sc   = smem;             // 16*1024 scores, then exp values
    float* pg   = sc + 16 * 1024;   // 16 gates
    float* red  = pg + 16;          // 16x8 reduction scratch
    float* rmax = red + 128;        // 16
    float* rsum = rmax + 16;        // 16

    const int bid = blockIdx.x;
    const int b   = bid >> 10;
    const int h   = (bid >> 6) & 15;
    const int q0  = (bid & 63) * 16;
    const unsigned tid  = threadIdx.x;
    const unsigned lane = tid & 31u;
    const unsigned wave = tid >> 5;
    const int half = (lane >> 4) & 1;
    const int nn   = lane & 15;

    if (tid < 16) pg[tid] = gate[((size_t)(b * 1024 + q0 + tid)) * 16 + h];
    __syncthreads();

    // q / tq fragments for this 16-row tile (K = DH = 64 -> two k-steps)
    const size_t qrow = ((size_t)b * 1024 + q0) * 1024 + h * 64;
    v16bf aq0 = load_frag_a(qs  + qrow,      1024, lane);
    v16bf aq1 = load_frag_a(qs  + qrow + 32, 1024, lane);
    v16bf at0 = load_frag_a(tqs + qrow,      1024, lane);
    v16bf at1 = load_frag_a(tqs + qrow + 32, 1024, lane);

    // ---- scores: each wave covers 16 of the 64 key tiles ----
    for (int kt = wave; kt < 64; kt += 4) {
        const size_t krow = ((size_t)b * 1024 + kt * 16) * 1024 + h * 64;
        v16bf bk0 = load_frag_b(ks  + krow,      1024, lane);
        v16bf bk1 = load_frag_b(ks  + krow + 32, 1024, lane);
        v16bf bt0 = load_frag_b(tks + krow,      1024, lane);
        v16bf bt1 = load_frag_b(tks + krow + 32, 1024, lane);
        v8f s = {}, t = {};
        s = __builtin_amdgcn_wmma_f32_16x16x32_bf16(false, aq0, false, bk0, (short)0, s, false, false);
        s = __builtin_amdgcn_wmma_f32_16x16x32_bf16(false, aq1, false, bk1, (short)0, s, false, false);
        t = __builtin_amdgcn_wmma_f32_16x16x32_bf16(false, at0, false, bt0, (short)0, t, false, false);
        t = __builtin_amdgcn_wmma_f32_16x16x32_bf16(false, at1, false, bt1, (short)0, t, false, false);
#pragma unroll
        for (int r = 0; r < 8; ++r) {
            const int m = r + half * 8;
            const float p = pg[m];
            float val = (1.0f - p) * s[r] + p * t[r];
            const unsigned char mk = mask[((size_t)b * 1024 + q0 + m) * 1024 + kt * 16 + nn];
            if (mk) val = -1e18f;
            sc[m * 1024 + kt * 16 + nn] = val;
        }
    }
    __syncthreads();

    // ---- softmax stats: 8 threads per row ----
    {
        const int row = tid >> 3;
        const int sub = tid & 7;
        float lm = -3.4e38f;
        for (int c = sub; c < 1024; c += 8) lm = fmaxf(lm, sc[row * 1024 + c]);
        red[row * 8 + sub] = lm;
        __syncthreads();
        if (sub == 0) {
            float m = red[row * 8];
            for (int i = 1; i < 8; ++i) m = fmaxf(m, red[row * 8 + i]);
            rmax[row] = m;
        }
        __syncthreads();
        const float mx = rmax[row];
        float ls = 0.0f;
        for (int c = sub; c < 1024; c += 8) {
            const float e = __expf(sc[row * 1024 + c] - mx);
            sc[row * 1024 + c] = e;
            ls += e;
        }
        red[row * 8 + sub] = ls;
        __syncthreads();
        if (sub == 0) {
            float ssum = 0.0f;
            for (int i = 0; i < 8; ++i) ssum += red[row * 8 + i];
            rsum[row] = ssum;
        }
    }
    __syncthreads();

    // ---- ctx = P @ V: wave owns one 16-wide d-tile; A-frag from LDS (cvt f32->bf16), B-frag from vT ----
    const int d0 = wave * 16;
    v8f acc = {};
    for (int kb = 0; kb < 1024; kb += 32) {
        const float* prow = sc + (lane & 15) * 1024 + kb + half * 8;
        v16bf ap;
#pragma unroll
        for (int i = 0; i < 8; ++i) {
            ap[i]     = (bf16)prow[i];
            ap[i + 8] = (bf16)prow[16 + i];
        }
        const size_t vrow = (((size_t)b * 16 + h) * 64 + d0) * 1024 + kb;
        v16bf bv = load_frag_b(vT + vrow, 1024, lane);
        acc = __builtin_amdgcn_wmma_f32_16x16x32_bf16(false, ap, false, bv, (short)0, acc, false, false);
    }
#pragma unroll
    for (int r = 0; r < 8; ++r) {
        const int m = r + half * 8;
        const float val = acc[r] / rsum[m];
        // write in [B, L, H, DH] layout so the reference "unshape" is a no-op
        ctx[((size_t)b * 1024 + q0 + m) * 1024 + h * 64 + d0 + nn] = (bf16)val;
    }
}

extern "C" void kernel_launch(void* const* d_in, const int* in_sizes, int n_in,
                              void* d_out, int out_size, void* d_ws, size_t ws_size,
                              hipStream_t stream) {
    const float* key   = (const float*)d_in[0];
    const float* value = (const float*)d_in[1];
    const float* query = (const float*)d_in[2];
    const unsigned char* mask = (const unsigned char*)d_in[3];  // jax bool -> 1 byte/elem
    const float* topic = (const float*)d_in[4];
    const float* Wq  = (const float*)d_in[5];  const float* bq  = (const float*)d_in[6];
    const float* Wk  = (const float*)d_in[7];  const float* bk  = (const float*)d_in[8];
    const float* Wv  = (const float*)d_in[9];  const float* bv  = (const float*)d_in[10];
    const float* Wtk = (const float*)d_in[11]; const float* btk = (const float*)d_in[12];
    const float* Wtv = (const float*)d_in[13]; const float* btv = (const float*)d_in[14];
    const float* Wtw = (const float*)d_in[15]; const float* btw = (const float*)d_in[16];
    const float* Wo  = (const float*)d_in[17]; const float* bo  = (const float*)d_in[18];
    float* out = (float*)d_out;

    const int BL = 4096, Dd = 1024;
    char* ws = (char*)d_ws;
    size_t cur = 0;
    auto alloc = [&](size_t bytes) -> char* {
        char* p = ws + cur;
        cur += (bytes + 255) & ~(size_t)255;
        return p;
    };

    bf16* qx    = (bf16*)alloc((size_t)BL * Dd * 2);   // query bf16
    bf16* kx    = (bf16*)alloc((size_t)BL * Dd * 2);   // key bf16
    bf16* vx    = (bf16*)alloc((size_t)BL * Dd * 2);   // value bf16
    bf16* tpad  = (bf16*)alloc((size_t)BL * 128 * 2);  // topic padded to 128
    bf16* wq    = (bf16*)alloc((size_t)Dd * Dd * 2);
    bf16* wk    = (bf16*)alloc((size_t)Dd * Dd * 2);
    bf16* wv    = (bf16*)alloc((size_t)Dd * Dd * 2);
    bf16* wtk   = (bf16*)alloc((size_t)Dd * Dd * 2);
    bf16* wo    = (bf16*)alloc((size_t)Dd * Dd * 2);
    bf16* wtv   = (bf16*)alloc((size_t)Dd * 128 * 2);  // Wtv padded to K=128
    float* qlin  = (float*)alloc((size_t)BL * Dd * 4);
    float* tklin = (float*)alloc((size_t)BL * Dd * 4);
    float* tvlin = (float*)alloc((size_t)BL * Dd * 4);
    bf16* q_s  = (bf16*)alloc((size_t)BL * Dd * 2);    // q_lin * 1/8, bf16
    bf16* k_s  = (bf16*)alloc((size_t)BL * Dd * 2);
    bf16* tk_s = (bf16*)alloc((size_t)BL * Dd * 2);
    bf16* tq_s = (bf16*)alloc((size_t)BL * Dd * 2);    // tv_lin * 1/8, bf16
    bf16* v_s  = (bf16*)alloc((size_t)BL * Dd * 2);
    bf16* vT   = (bf16*)alloc((size_t)BL * Dd * 2);    // [B,H,DH,L]
    float* gate = (float*)alloc((size_t)BL * 16 * 4);
    bf16* ctx  = (bf16*)alloc((size_t)BL * Dd * 2);
    (void)ws_size; (void)n_in; (void)in_sizes; (void)out_size;

    const int NELEM = BL * Dd;                    // 4,194,304
    cvt_f32_bf16<<<(NELEM + 255) / 256, 256, 0, stream>>>(query, qx, NELEM);
    cvt_f32_bf16<<<(NELEM + 255) / 256, 256, 0, stream>>>(key,   kx, NELEM);
    cvt_f32_bf16<<<(NELEM + 255) / 256, 256, 0, stream>>>(value, vx, NELEM);
    const int WEL = Dd * Dd;
    cvt_f32_bf16<<<(WEL + 255) / 256, 256, 0, stream>>>(Wq,  wq,  WEL);
    cvt_f32_bf16<<<(WEL + 255) / 256, 256, 0, stream>>>(Wk,  wk,  WEL);
    cvt_f32_bf16<<<(WEL + 255) / 256, 256, 0, stream>>>(Wv,  wv,  WEL);
    cvt_f32_bf16<<<(WEL + 255) / 256, 256, 0, stream>>>(Wtk, wtk, WEL);
    cvt_f32_bf16<<<(WEL + 255) / 256, 256, 0, stream>>>(Wo,  wo,  WEL);
    pad_cvt<<<(BL * 128 + 255) / 256, 256, 0, stream>>>(topic, tpad, BL, 100, 128);
    pad_cvt<<<(Dd * 128 + 255) / 256, 256, 0, stream>>>(Wtv,   wtv,  Dd, 100, 128);

    dim3 gblock(256);
    dim3 ggrid(BL / 32, Dd / 512);                     // (128, 2)
    const size_t shA_K1024 = (size_t)32 * 1024 * 2;    // 64 KB
    const size_t shA_K128  = (size_t)32 * 128  * 2;    // 8 KB
    gemm_nt_bf16<<<ggrid, gblock, shA_K1024, stream>>>(qx,   wq,  bq,  qlin,   q_s,  0.125f, BL, Dd, Dd);
    gemm_nt_bf16<<<ggrid, gblock, shA_K1024, stream>>>(kx,   wk,  bk,  nullptr, k_s,  1.0f,   BL, Dd, Dd);
    gemm_nt_bf16<<<ggrid, gblock, shA_K1024, stream>>>(vx,   wv,  bv,  nullptr, v_s,  1.0f,   BL, Dd, Dd);
    gemm_nt_bf16<<<ggrid, gblock, shA_K1024, stream>>>(kx,   wtk, btk, tklin,  tk_s, 1.0f,   BL, Dd, Dd);
    gemm_nt_bf16<<<ggrid, gblock, shA_K128,  stream>>>(tpad, wtv, btv, tvlin,  tq_s, 0.125f, BL, Dd, 128);

    transpose_v<<<(NELEM + 255) / 256, 256, 0, stream>>>(v_s, vT);
    gate_kernel<<<(BL * 16 + 255) / 256, 256, 0, stream>>>(qlin, tklin, tvlin, Wtw, btw, gate);

    const size_t shbytes = (size_t)(16 * 1024 + 16 + 128 + 16 + 16) * sizeof(float);
    attn_kernel<<<4096, 128, shbytes, stream>>>(q_s, k_s, tq_s, tk_s, vT, gate, mask, ctx);

    gemm_nt_bf16<<<ggrid, gblock, shA_K1024, stream>>>(ctx, wo, bo, out, nullptr, 1.0f, BL, Dd, Dd);
}
